// LiquidPerceptionUnit_68685116997889
// MI455X (gfx1250) — matrix operand order repaired and checked by reading, not compiled, not run on hardware
//
#include <hip/hip_runtime.h>
#include <hip/hip_bf16.h>
#include <math.h>

// ---------------------------------------------------------------------------
// LiquidPerceptionUnit on gfx1250 (MI455X, wave32, WMMA)
//
// Phase 1: encoder_kernel -> pooled (16384 x 128) f16   (parallel, cheap)
// Phase 2: fc_gemm_kernel -> feats (T,B,128) f16 via v_wmma_f32_16x16x32_f16
// Phase 3: scan_kernel    -> 512 sequential CfC steps, 2 blocks x 16 rows,
//          all 12 per-layer GEMMs on WMMA; packed f16 weights stream from L2;
//          next-step feature slab double-buffered in registers so its L2
//          latency overlaps the current step's compute.
// ---------------------------------------------------------------------------

typedef __attribute__((ext_vector_type(16))) _Float16 v16h;
typedef __attribute__((ext_vector_type(8)))  _Float16 v8h;
typedef __attribute__((ext_vector_type(8)))  float    v8f;

union Frag16 { v16h v; v8h h[2]; };

// ---------------------------------------------------------------------------
// Pack W (hid x cat, row-major) into WMMA B-fragment layout, f16, masked,
// zero-padded to Kp x Np.  B[k][n] = W[n][k] * mask[n][k].
// Fragment layout (16-bit B, 32K x 16N tile, wave32):
//   lane = 16*(k%32/16) + n%16 ; j = k%16 ; idx = (((kt*Nt+nt)*32+lane)*16+j)
// ---------------------------------------------------------------------------
__global__ void pack_b_kernel(const float* __restrict__ W,
                              const float* __restrict__ mask,
                              int hid, int cat, int Kp, int Np,
                              _Float16* __restrict__ dst)
{
    int idx = blockIdx.x * blockDim.x + threadIdx.x;
    int tot = Kp * Np;
    if (idx >= tot) return;
    int k = idx / Np, n = idx % Np;
    float v = 0.f;
    if (k < cat && n < hid) {
        v = W[(size_t)n * cat + k];
        if (mask) v *= mask[(size_t)n * cat + k];
    }
    int kt = k >> 5, rem = k & 31, lg = rem >> 4, j = rem & 15;
    int nt = n >> 4, lane = (lg << 4) | (n & 15);
    int Nt = Np >> 4;
    dst[((((size_t)kt * Nt + nt) * 32 + lane) << 4) + j] = (_Float16)v;
}

// Pad the 4 per-layer bias vectors to stride 128 with zeros (padded WMMA
// output columns then produce exactly h=0 through the CfC combine).
__global__ void pad_bias_kernel(const float* __restrict__ b1, const float* __restrict__ b2,
                                const float* __restrict__ ba, const float* __restrict__ bb,
                                int hid, float* __restrict__ dst)
{
    int idx = blockIdx.x * blockDim.x + threadIdx.x;
    if (idx >= 512) return;
    int mat = idx >> 7, n = idx & 127;
    const float* s = (mat == 0) ? b1 : (mat == 1) ? b2 : (mat == 2) ? ba : bb;
    dst[idx] = (n < hid) ? s[n] : 0.f;
}

// ---------------------------------------------------------------------------
// CNN encoder: one wave per sample, 8 samples per 256-thread block.
// ---------------------------------------------------------------------------
__global__ __launch_bounds__(256) void encoder_kernel(
    const float* __restrict__ x,
    const float* __restrict__ c1w, const float* __restrict__ c1b,
    const float* __restrict__ c2w, const float* __restrict__ c2b,
    _Float16* __restrict__ pooled)
{
    __shared__ float xs[8][258];
    __shared__ float p1s[8][16][10];
    __shared__ float c2ws[1536];

    const int tid  = threadIdx.x;
    const int w    = tid >> 5, lane = tid & 31;
    const int s    = blockIdx.x * 8 + w;

    for (int i = tid; i < 1536; i += 256) c2ws[i] = c2w[i];

    const float* xp = x + (size_t)s * 256;
    for (int i = 0; i < 8; ++i) xs[w][1 + lane + 32 * i] = xp[lane + 32 * i];
    if (lane == 0) { xs[w][0] = 0.f; xs[w][257] = 0.f; }
    if (lane < 16) { p1s[w][lane][0] = 0.f; p1s[w][lane][9] = 0.f; }
    __syncthreads();

    // conv1 + relu + mean-pool-32
    for (int u = 0; u < 4; ++u) {
        int idx = lane + 32 * u;
        int c = idx >> 3, j = idx & 7;
        float w0 = c1w[c*3+0], w1 = c1w[c*3+1], w2 = c1w[c*3+2], bb = c1b[c];
        float acc = 0.f;
        int base = 32 * j;
        for (int p = 0; p < 32; ++p) {
            float v = bb + w0 * xs[w][base+p] + w1 * xs[w][base+p+1] + w2 * xs[w][base+p+2];
            acc += fmaxf(v, 0.f);
        }
        p1s[w][c][1 + j] = acc * (1.f / 32.f);
    }
    __syncthreads();

    // conv2 + relu + mean-pool-2
    for (int u = 0; u < 4; ++u) {
        int idx = lane + 32 * u;
        int c = idx >> 2, j = idx & 3;
        float bb = c2b[c];
        float acc = 0.f;
        for (int half = 0; half < 2; ++half) {
            int ii = 2 * j + half;
            float v = bb;
            for (int ci = 0; ci < 16; ++ci) {
                const float* wp = &c2ws[(c * 16 + ci) * 3];
                v += wp[0]*p1s[w][ci][ii] + wp[1]*p1s[w][ci][ii+1] + wp[2]*p1s[w][ci][ii+2];
            }
            acc += fmaxf(v, 0.f);
        }
        pooled[(size_t)s * 128 + idx] = (_Float16)(acc * 0.5f);
    }
}

// ---------------------------------------------------------------------------
// feats = pooled @ fc_w.T + fc_b  -- (16384 x 128 x 128) GEMM on WMMA.
// ---------------------------------------------------------------------------
__global__ __launch_bounds__(256) void fc_gemm_kernel(
    const _Float16* __restrict__ pooled,
    const _Float16* __restrict__ bpack,
    const float* __restrict__ fcb,
    _Float16* __restrict__ feats)
{
    const int lane = threadIdx.x & 31;
    const int tidw = blockIdx.x * 8 + (threadIdx.x >> 5);
    const int mt = tidw >> 3, nt = tidw & 7;

    v8f acc = {};
    for (int kt = 0; kt < 4; ++kt) {
        int m  = mt * 16 + (lane & 15);
        int kb = kt * 32 + ((lane >> 4) << 4);
        const v8h* ap = (const v8h*)(pooled + (size_t)m * 128 + kb);
        Frag16 a; a.h[0] = ap[0]; a.h[1] = ap[1];
        const v8h* bp = (const v8h*)(bpack + ((((size_t)kt * 8 + nt) * 32 + lane) << 4));
        Frag16 b; b.h[0] = bp[0]; b.h[1] = bp[1];
        acc = __builtin_amdgcn_wmma_f32_16x16x32_f16(false, a.v, false, b.v,
                                                     (short)0, acc, false, false);
    }
    int n = nt * 16 + (lane & 15);
    float bias = fcb[n];
#pragma unroll
    for (int r = 0; r < 8; ++r) {
        int m = mt * 16 + r + ((lane >> 4) << 3);
        int bb = m >> 9, t = m & 511;                 // sample = b*T + t
        feats[((size_t)t * 32 + bb) * 128 + n] = (_Float16)(acc[r] + bias);
    }
}

// ---------------------------------------------------------------------------
// One CfC layer step: 4 WMMA accumulators per output tile (w1,w2,wa,wb),
// f32 activation math, f16 hidden tile to LDS (motor layer also to global).
// Biases are hoisted by the caller into registers (loop-invariant).
// ---------------------------------------------------------------------------
template<int KP, int NT, int KT>
__device__ __forceinline__ void cfc_layer(
    const _Float16* xbuf,                 // LDS, [16][KP]
    const _Float16* __restrict__ pack,    // 4 matrices, each KP*NP halves
    float bb1, float bb2, float bba, float bbb,
    _Float16* hout,                       // LDS, [16][NT*16]
    float* __restrict__ gout, int t)      // motor output base or nullptr
{
    const int w    = threadIdx.x >> 5;    // wave-uniform -> EXEC stays all-1s
    const int lane = threadIdx.x & 31;
    if (w >= NT) return;
    const int nt = w;
    constexpr int NP = NT * 16;
    constexpr size_t MS = (size_t)KP * NP;

    v8f a0 = {}, a1 = {}, a2 = {}, a3 = {};
    for (int kt = 0; kt < KT; ++kt) {
        int m  = lane & 15;
        int kb = kt * 32 + ((lane >> 4) << 4);
        const v8h* ap = (const v8h*)(xbuf + m * KP + kb);
        Frag16 a; a.h[0] = ap[0]; a.h[1] = ap[1];

        size_t boff = (((size_t)kt * NT + nt) * 32 + lane) << 4;
        Frag16 b0, b1, b2, b3;
        const v8h* q0 = (const v8h*)(pack + boff);
        const v8h* q1 = (const v8h*)(pack + MS + boff);
        const v8h* q2 = (const v8h*)(pack + 2 * MS + boff);
        const v8h* q3 = (const v8h*)(pack + 3 * MS + boff);
        b0.h[0] = q0[0]; b0.h[1] = q0[1];
        b1.h[0] = q1[0]; b1.h[1] = q1[1];
        b2.h[0] = q2[0]; b2.h[1] = q2[1];
        b3.h[0] = q3[0]; b3.h[1] = q3[1];

        a0 = __builtin_amdgcn_wmma_f32_16x16x32_f16(false, a.v, false, b0.v, (short)0, a0, false, false);
        a1 = __builtin_amdgcn_wmma_f32_16x16x32_f16(false, a.v, false, b1.v, (short)0, a1, false, false);
        a2 = __builtin_amdgcn_wmma_f32_16x16x32_f16(false, a.v, false, b2.v, (short)0, a2, false, false);
        a3 = __builtin_amdgcn_wmma_f32_16x16x32_f16(false, a.v, false, b3.v, (short)0, a3, false, false);
    }

    int n = nt * 16 + (lane & 15);
#pragma unroll
    for (int r = 0; r < 8; ++r) {
        int m = r + ((lane >> 4) << 3);
        float f1   = tanhf(a0[r] + bb1);
        float f2   = tanhf(a1[r] + bb2);
        float sarg = (a2[r] + bba) + (a3[r] + bbb);
        float ti   = 1.f / (1.f + __expf(-sarg));
        float hv   = f1 * (1.f - ti) + ti * f2;
        hout[m * NP + n] = (_Float16)hv;
        if (gout) gout[((size_t)m * 512 + t) * 64 + n] = hv;
    }
}

// ---------------------------------------------------------------------------
// Sequential scan: 2 blocks x 16 batch rows (rows are recurrence-independent),
// 256 threads = 8 waves per block. Hidden state in LDS (f16). The feature
// slab for step t+1 is fetched into registers during step t so its L2
// latency is off the critical path.
// ---------------------------------------------------------------------------
__global__ __launch_bounds__(256) void scan_kernel(
    const _Float16* __restrict__ feats,
    const _Float16* __restrict__ p0,
    const _Float16* __restrict__ p1,
    const _Float16* __restrict__ p2,
    const float* __restrict__ biasAll,
    float* __restrict__ out)
{
    __shared__ __attribute__((aligned(16))) _Float16 xbuf[16 * 256];
    __shared__ __attribute__((aligned(16))) _Float16 h0[16 * 128];
    __shared__ __attribute__((aligned(16))) _Float16 h1[16 * 80];
    __shared__ __attribute__((aligned(16))) _Float16 h2[16 * 64];

    const int tid  = threadIdx.x;
    const int row0 = blockIdx.x * 16;
    const int w    = tid >> 5, lane = tid & 31;

    for (int i = tid; i < 16 * 128; i += 256) h0[i] = (_Float16)0.f;
    for (int i = tid; i < 16 * 80;  i += 256) h1[i] = (_Float16)0.f;
    for (int i = tid; i < 16 * 64;  i += 256) h2[i] = (_Float16)0.f;

    // Hoist loop-invariant biases into registers (per wave tile column n).
    const int nn = w * 16 + (lane & 15);
    float b0v0 = 0.f, b0v1 = 0.f, b0v2 = 0.f, b0v3 = 0.f;
    float b1v0 = 0.f, b1v1 = 0.f, b1v2 = 0.f, b1v3 = 0.f;
    float b2v0 = 0.f, b2v1 = 0.f, b2v2 = 0.f, b2v3 = 0.f;
    if (w < 8) { b0v0 = biasAll[nn];        b0v1 = biasAll[128 + nn];
                 b0v2 = biasAll[256 + nn];  b0v3 = biasAll[384 + nn]; }
    if (w < 5) { b1v0 = biasAll[512 + nn];  b1v1 = biasAll[640 + nn];
                 b1v2 = biasAll[768 + nn];  b1v3 = biasAll[896 + nn]; }
    if (w < 4) { b2v0 = biasAll[1024 + nn]; b2v1 = biasAll[1152 + nn];
                 b2v2 = biasAll[1280 + nn]; b2v3 = biasAll[1408 + nn]; }

    float* motor = out + (size_t)row0 * 512 * 64;

    // Register double-buffer of the feature slab (16 rows x 128 halves):
    // thread tid owns the contiguous 32B chunk at flat offset tid*16.
    const int fm = tid >> 3;                  // slab row this thread owns
    const int fc = (tid & 7) << 4;            // column start within row
    v8h fa, fb;
    {
        const v8h* s0 = (const v8h*)(feats + ((size_t)0 * 32 + row0) * 128 + tid * 16);
        fa = s0[0]; fb = s0[1];
    }
    __syncthreads();

    for (int t = 0; t < 512; ++t) {
        // ---- layer 0: xx = [feat(128) | h0(116+pad)] -> KP=256 ----
        {   // feature half (from registers, loaded one step ahead)
            v8h* xd = (v8h*)(xbuf + fm * 256 + fc);
            xd[0] = fa; xd[1] = fb;
            // h0 half (aligned LDS->LDS vector copy)
            const v8h* hs = (const v8h*)(h0 + fm * 128 + fc);
            v8h* xh = (v8h*)(xbuf + fm * 256 + 128 + fc);
            xh[0] = hs[0]; xh[1] = hs[1];
        }
        // issue next step's slab fetch; overlaps this whole step's compute
        {
            int tn = (t + 1 < 512) ? t + 1 : 0;
            const v8h* sn = (const v8h*)(feats + ((size_t)tn * 32 + row0) * 128 + tid * 16);
            fa = sn[0]; fb = sn[1];
        }
        __syncthreads();
        cfc_layer<256, 8, 8>(xbuf, p0, b0v0, b0v1, b0v2, b0v3, h0, nullptr, t);
        __syncthreads();

        // ---- layer 1: xx = [h0(116) | h1(76)] -> KP=192 ----
        for (int i = 0; i < 12; ++i) {
            int e = tid + 256 * i;
            int m = e / 192, c = e % 192;
            xbuf[e] = (c < 116) ? h0[m * 128 + c] : h1[m * 80 + (c - 116)];
        }
        __syncthreads();
        cfc_layer<192, 5, 6>(xbuf, p1, b1v0, b1v1, b1v2, b1v3, h1, nullptr, t);
        __syncthreads();

        // ---- layer 2: xx = [h1(76) | h2(64) | pad] -> KP=160 ----
        for (int i = 0; i < 10; ++i) {
            int e = tid + 256 * i;
            int m = e / 160, c = e % 160;
            _Float16 v = (_Float16)0.f;
            if (c < 76)       v = h1[m * 80 + c];
            else if (c < 140) v = h2[m * 64 + (c - 76)];
            xbuf[e] = v;
        }
        __syncthreads();
        cfc_layer<160, 4, 5>(xbuf, p2, b2v0, b2v1, b2v2, b2v3, h2, motor, t);
        __syncthreads();
    }

    // new_state = concat(h0[:116], h1[:76], h2[:64]) per row, f32
    float* state = out + (size_t)32 * 512 * 64;
    for (int i = 0; i < 16; ++i) {
        int e = tid + 256 * i;
        int m = e >> 8, c = e & 255;
        float v;
        if (c < 116)      v = (float)h0[m * 128 + c];
        else if (c < 192) v = (float)h1[m * 80 + (c - 116)];
        else              v = (float)h2[m * 64 + (c - 192)];
        state[(size_t)(row0 + m) * 256 + c] = v;
    }
}

// ---------------------------------------------------------------------------
extern "C" void kernel_launch(void* const* d_in, const int* in_sizes, int n_in,
                              void* d_out, int out_size, void* d_ws, size_t ws_size,
                              hipStream_t stream)
{
    (void)in_sizes; (void)n_in; (void)out_size; (void)ws_size;

    const float* x   = (const float*)d_in[0];
    const float* c1w = (const float*)d_in[1];
    const float* c1b = (const float*)d_in[2];
    const float* c2w = (const float*)d_in[3];
    const float* c2b = (const float*)d_in[4];
    const float* fcw = (const float*)d_in[5];
    const float* fcb = (const float*)d_in[6];

    char* ws = (char*)d_ws;                         // ~8.9 MB total
    _Float16* pooled  = (_Float16*)(ws + 0);        // 16384*128 f16 = 4 MB
    _Float16* feats   = (_Float16*)(ws + 4194304);  // 4 MB
    _Float16* fcpack  = (_Float16*)(ws + 8388608);  // 32 KB
    _Float16* l0pack  = (_Float16*)(ws + 8421376);  // 256 KB
    _Float16* l1pack  = (_Float16*)(ws + 8683520);  // 120 KB
    _Float16* l2pack  = (_Float16*)(ws + 8806400);  // 80 KB
    float*    biasAll = (float*)   (ws + 8888320);  // 3*512 f32

    pack_b_kernel<<<(128 * 128 + 255) / 256, 256, 0, stream>>>(
        fcw, nullptr, 128, 128, 128, 128, fcpack);

    const int hidA[3] = {116, 76, 64};
    const int catA[3] = {244, 192, 140};
    const int KpA[3]  = {256, 192, 160};
    const int NpA[3]  = {128, 80, 64};
    _Float16* packA[3] = {l0pack, l1pack, l2pack};

    for (int l = 0; l < 3; ++l) {
        int base = 7 + l * 9;                       // mask, w1,w2,wa,wb, b1,b2,ba,bb
        const float* mask = (const float*)d_in[base + 0];
        int KN = KpA[l] * NpA[l];
        for (int mmat = 0; mmat < 4; ++mmat) {
            const float* W  = (const float*)d_in[base + 1 + mmat];
            const float* mk = (mmat < 2) ? mask : nullptr;  // mask on w1,w2 only
            pack_b_kernel<<<(KN + 255) / 256, 256, 0, stream>>>(
                W, mk, hidA[l], catA[l], KpA[l], NpA[l],
                packA[l] + (size_t)mmat * KN);
        }
        pad_bias_kernel<<<2, 256, 0, stream>>>(
            (const float*)d_in[base + 5], (const float*)d_in[base + 6],
            (const float*)d_in[base + 7], (const float*)d_in[base + 8],
            hidA[l], biasAll + l * 512);
    }

    encoder_kernel<<<2048, 256, 0, stream>>>(x, c1w, c1b, c2w, c2b, pooled);
    fc_gemm_kernel<<<1024, 256, 0, stream>>>(pooled, fcpack, fcb, feats);
    scan_kernel<<<2, 256, 0, stream>>>(feats, l0pack, l1pack, l2pack, biasAll,
                                       (float*)d_out);
}
